// FusionTransformerModel_17789754540181
// MI455X (gfx1250) — matrix operand order repaired and checked by reading, not compiled, not run on hardware
//
#include <hip/hip_runtime.h>
#include <hip/hip_bf16.h>
#include <math.h>
#include <stdint.h>

typedef __attribute__((ext_vector_type(16))) _Float16 v16h;
typedef __attribute__((ext_vector_type(8)))  _Float16 v8h;
typedef __attribute__((ext_vector_type(4)))  _Float16 v4h;
typedef __attribute__((ext_vector_type(8)))  float    v8f;
typedef __attribute__((ext_vector_type(4)))  unsigned int v4u;
typedef __attribute__((ext_vector_type(8)))  int      v8i;
typedef __attribute__((ext_vector_type(4)))  int      v4i;

constexpr int kD   = 1024;
constexpr int kDFF = 4096;
constexpr int kNH  = 16;
constexpr int kHD  = 64;
constexpr int kB   = 2;
constexpr int kSD  = 1024;
constexpr int kST  = 1024;
constexpr int kS   = kSD + kST;   // 2048
constexpr int kTOK = kB * kS;     // 4096

// ---------------------------------------------------------------------------
// Tensor Data Mover: 2D f16 tile (tile_d1 rows x tile_d0 elems) from global
// (row stride = stride_e elements) into LDS at lds_off, contiguous rows.
// D# layout per CDNA5 ISA (group0 128b, group1 256b).
// Must be called wave-uniformly (TDM ignores EXEC; issue from one wave).
// ---------------------------------------------------------------------------
__device__ __forceinline__ void tdm_load_2d(const _Float16* gptr,
                                            unsigned lds_off,
                                            int stride_e,
                                            int tile_d0, int tile_d1) {
  const unsigned long long ga = (unsigned long long)(uintptr_t)gptr;
  v4u g0;
  g0[0] = 1u;                                            // count=1, user mode
  g0[1] = lds_off;                                       // lds_addr
  g0[2] = (unsigned)(ga & 0xFFFFFFFFu);                  // global_addr[31:0]
  g0[3] = (unsigned)((ga >> 32) & 0x01FFFFFFu)           // global_addr[56:32]
        | (2u << 30);                                    // type=2 (image)
  v8i g1;
  g1[0] = (int)(1u << 16);                               // data_size=1 -> 2B
  g1[1] = (int)((unsigned)(tile_d0 & 0xFFFF) << 16);     // tensor_dim0[15:0]
  g1[2] = (int)((((unsigned)tile_d0 >> 16) & 0xFFFFu)    // tensor_dim0[31:16]
        | ((unsigned)(tile_d1 & 0xFFFF) << 16));         // tensor_dim1[15:0]
  g1[3] = (int)((((unsigned)tile_d1 >> 16) & 0xFFFFu)    // tensor_dim1[31:16]
        | ((unsigned)tile_d0 << 16));                    // tile_dim0
  g1[4] = (int)((unsigned)tile_d1 & 0xFFFFu);            // tile_dim1 (dim2=0)
  g1[5] = stride_e;                                      // dim0_stride[31:0]
  g1[6] = 0;                                             // dim0_stride[47:32]
  g1[7] = 0;
  const v4i z4 = {0, 0, 0, 0};
#if __clang_major__ >= 23
  const v8i z8 = {0, 0, 0, 0, 0, 0, 0, 0};
  __builtin_amdgcn_tensor_load_to_lds(g0, g1, z4, z4, z8, 0);
#else
  __builtin_amdgcn_tensor_load_to_lds(g0, g1, z4, z4, 0);
#endif
}

// ---------------------------------------------------------------------------
__device__ __forceinline__ float block_sum_256(float v, float* red) {
  const int tid = threadIdx.x;
  #pragma unroll
  for (int off = 16; off > 0; off >>= 1) v += __shfl_xor(v, off, 32);
  __syncthreads();
  if ((tid & 31) == 0) red[tid >> 5] = v;
  __syncthreads();
  float tot = 0.f;
  #pragma unroll
  for (int i = 0; i < 8; ++i) tot += red[i];
  return tot;
}

__device__ __forceinline__ bool token_valid(int s, int ld, int lt) {
  return (s < kSD) ? (s < ld) : ((s - kSD) < lt);
}

// ---------------------------------------------------------------------------
// Weight conversion f32 -> f16 (count divisible by 1024)
// ---------------------------------------------------------------------------
__global__ void __launch_bounds__(256)
f32_to_f16_kernel(const float* __restrict__ src, _Float16* __restrict__ dst) {
  const size_t i = ((size_t)blockIdx.x * 256 + threadIdx.x) * 4;
  const float4 f = *(const float4*)(src + i);
  v4h h;
  h[0] = (_Float16)f.x; h[1] = (_Float16)f.y;
  h[2] = (_Float16)f.z; h[3] = (_Float16)f.w;
  *(v4h*)(dst + i) = h;
}

// ---------------------------------------------------------------------------
// Embedding: transpose-in, type emb, LayerNorm, mask. Writes X f32 + Xh f16.
// ---------------------------------------------------------------------------
__global__ void __launch_bounds__(256)
embed_kernel(const float* __restrict__ xd, const float* __restrict__ xt,
             const float* __restrict__ temb,
             const float* __restrict__ lw, const float* __restrict__ lb,
             const int* __restrict__ lend, const int* __restrict__ lent,
             float* __restrict__ X, _Float16* __restrict__ Xh) {
  __shared__ float red[8];
  const int t = blockIdx.x;
  const int b = t / kS;
  const int s = t % kS;
  const float* src;
  const float* te;
  if (s < kSD) { src = xd + ((size_t)s * kB + b) * kD;         te = temb; }
  else         { src = xt + ((size_t)(s - kSD) * kB + b) * kD; te = temb + kD; }

  float vals[4];
  float sum = 0.f;
  #pragma unroll
  for (int i = 0; i < 4; ++i) {
    const int d = threadIdx.x + i * 256;
    vals[i] = src[d] + te[d];
    sum += vals[i];
  }
  const float mean = block_sum_256(sum, red) * (1.0f / kD);
  float vs = 0.f;
  #pragma unroll
  for (int i = 0; i < 4; ++i) { const float c = vals[i] - mean; vs += c * c; }
  const float var = block_sum_256(vs, red) * (1.0f / kD);
  const float inv = rsqrtf(var + 1e-12f);
  const float msk = token_valid(s, lend[b], lent[b]) ? 1.0f : 0.0f;
  #pragma unroll
  for (int i = 0; i < 4; ++i) {
    const int d = threadIdx.x + i * 256;
    const float o = ((vals[i] - mean) * inv * lw[d] + lb[d]) * msk;
    X[(size_t)t * kD + d]  = o;
    Xh[(size_t)t * kD + d] = (_Float16)o;
  }
}

// ---------------------------------------------------------------------------
// GEMM: C[M,N] = A[M,K](f16) @ W[N,K](f16)^T + bias
// TDM double-buffered LDS staging; 8 waves x (16M x 128N); K-step 64.
// MODE 0: f32 out | 1: f16 out | 2: f16 out + exact GELU
// ---------------------------------------------------------------------------
template<int MODE>
__global__ void __launch_bounds__(256)
gemm_wmma(const _Float16* __restrict__ A, const _Float16* __restrict__ W,
          const float* __restrict__ bias, void* __restrict__ Cout,
          int M, int N, int K) {
  __shared__ __align__(16) _Float16 As[2][128 * 64];
  __shared__ __align__(16) _Float16 Bs[2][128 * 64];
  const int tid   = threadIdx.x;
  const int lane  = tid & 31;
  const int wv    = tid >> 5;
  const int lrow  = lane & 15;
  const int lhalf = lane >> 4;
  const int m0 = blockIdx.y * 128;
  const int n0 = blockIdx.x * 128;

  v8f acc[8] = {};

  const unsigned ldsA[2] = { (unsigned)(uintptr_t)&As[0][0],
                             (unsigned)(uintptr_t)&As[1][0] };
  const unsigned ldsB[2] = { (unsigned)(uintptr_t)&Bs[0][0],
                             (unsigned)(uintptr_t)&Bs[1][0] };
  const _Float16* gA = A + (size_t)m0 * K;
  const _Float16* gW = W + (size_t)n0 * K;

  if (wv == 0) {
    tdm_load_2d(gA, ldsA[0], K, 64, 128);
    tdm_load_2d(gW, ldsB[0], K, 64, 128);
  }

  for (int k0 = 0; k0 < K; k0 += 64) {
    const int buf = (k0 >> 6) & 1;
    if (wv == 0) {
      const int kn = k0 + 64;
      if (kn < K) {
        tdm_load_2d(gA + kn, ldsA[buf ^ 1], K, 64, 128);
        tdm_load_2d(gW + kn, ldsB[buf ^ 1], K, 64, 128);
        __builtin_amdgcn_s_wait_tensorcnt(2);   // current buffer complete
      } else {
        __builtin_amdgcn_s_wait_tensorcnt(0);
      }
    }
    __syncthreads();

    const _Float16* asb = &As[buf][0];
    const _Float16* bsb = &Bs[buf][0];
    #pragma unroll
    for (int ks = 0; ks < 2; ++ks) {
      const _Float16* ap = asb + (wv * 16 + lrow) * 64 + ks * 32 + lhalf * 16;
      const v8h a0 = *(const v8h*)(ap);
      const v8h a1 = *(const v8h*)(ap + 8);
      const v16h af = __builtin_shufflevector(a0, a1,
          0, 1, 2, 3, 4, 5, 6, 7, 8, 9, 10, 11, 12, 13, 14, 15);
      #pragma unroll
      for (int nt = 0; nt < 8; ++nt) {
        const v16h bf = *(const v16h*)(bsb + (nt * 16 + lrow) * 64
                                       + ks * 32 + lhalf * 16);
        acc[nt] = __builtin_amdgcn_wmma_f32_16x16x32_f16(
            false, af, false, bf, (short)0, acc[nt], false, false);
      }
    }
    __syncthreads();
  }

  #pragma unroll
  for (int nt = 0; nt < 8; ++nt) {
    const int col = n0 + nt * 16 + lrow;
    const float bb = bias[col];
    #pragma unroll
    for (int r = 0; r < 8; ++r) {
      const int row = m0 + wv * 16 + r + lhalf * 8;
      float v = acc[nt][r] + bb;
      if (MODE == 2) v = 0.5f * v * (1.0f + erff(v * 0.70710678118654752f));
      if (MODE == 0) ((float*)Cout)[(size_t)row * N + col] = v;
      else ((_Float16*)Cout)[(size_t)row * N + col] = (_Float16)v;
    }
  }
}

// ---------------------------------------------------------------------------
// Flash attention over f16 Q/K/V; Q + K tiles via TDM, V staged transposed.
// Block: 128 threads (4 waves), 64-query tile; key tiles of 32.
// Softmax normalizer accumulated with WMMA (P @ ones) instead of shuffles.
// ---------------------------------------------------------------------------
__global__ void __launch_bounds__(128)
attention_kernel(const _Float16* __restrict__ Qh, const _Float16* __restrict__ Kh,
                 const _Float16* __restrict__ Vh, _Float16* __restrict__ Oh,
                 const int* __restrict__ lend, const int* __restrict__ lent) {
  __shared__ __align__(16) _Float16 Qs[64 * 64];    // [q][d]
  __shared__ __align__(16) _Float16 Ks[32 * 64];    // [key][d]
  __shared__ __align__(16) _Float16 Vt[64 * 32];    // [d][key]
  __shared__ __align__(16) _Float16 Ps[4][16 * 32]; // per-wave [q][key]

  const int tid   = threadIdx.x;
  const int lane  = tid & 31;
  const int wv    = tid >> 5;
  const int lrow  = lane & 15;
  const int lhalf = lane >> 4;
  const int q0 = blockIdx.x * 64;
  const int h  = blockIdx.y;
  const int b  = blockIdx.z;
  const int ld = lend[b], lt = lent[b];
  const float scale = 0.125f;

  const size_t base = (size_t)b * kS * kD + (size_t)h * kHD;
  const unsigned ldsQ = (unsigned)(uintptr_t)&Qs[0];
  const unsigned ldsK = (unsigned)(uintptr_t)&Ks[0];

  if (wv == 0) {
    tdm_load_2d(Qh + base + (size_t)q0 * kD, ldsQ, kD, 64, 64);
    __builtin_amdgcn_s_wait_tensorcnt(0);
  }
  __syncthreads();

  v16h qf[2];
  #pragma unroll
  for (int ks = 0; ks < 2; ++ks) {
    const _Float16* qp = &Qs[(wv * 16 + lrow) * 64 + ks * 32 + lhalf * 16];
    const v8h a0 = *(const v8h*)(qp);
    const v8h a1 = *(const v8h*)(qp + 8);
    qf[ks] = __builtin_shufflevector(a0, a1,
        0, 1, 2, 3, 4, 5, 6, 7, 8, 9, 10, 11, 12, 13, 14, 15);
  }
  __syncthreads();

  // all-ones B fragment: lacc = P @ ones gives per-row softmax normalizer
  v16h onesf;
  #pragma unroll
  for (int i = 0; i < 16; ++i) onesf[i] = (_Float16)1.0f;

  v8f o[4] = {};
  v8f lacc = {};
  float mrow[8];
  #pragma unroll
  for (int r = 0; r < 8; ++r) mrow[r] = -3.0e38f;

  for (int kt = 0; kt < kS / 32; ++kt) {
    // K tile via TDM (wave 0) while all threads stage V transposed
    if (wv == 0) tdm_load_2d(Kh + base + (size_t)(kt * 32) * kD, ldsK, kD, 64, 32);
    {
      const int r  = tid >> 2;          // key 0..31
      const int c0 = (tid & 3) * 16;    // d offset
      const _Float16* gv = Vh + base + (size_t)(kt * 32 + r) * kD + c0;
      const v8h f0 = *(const v8h*)(gv);
      const v8h f1 = *(const v8h*)(gv + 8);
      #pragma unroll
      for (int j = 0; j < 8; ++j) {
        Vt[(c0 + j) * 32 + r]     = f0[j];
        Vt[(c0 + 8 + j) * 32 + r] = f1[j];
      }
    }
    if (wv == 0) __builtin_amdgcn_s_wait_tensorcnt(0);
    __syncthreads();

    // scores: 16q x 32keys
    v8f sc0 = {}, sc1 = {};
    #pragma unroll
    for (int ks = 0; ks < 2; ++ks) {
      const v16h b0 = *(const v16h*)&Ks[(lrow) * 64 + ks * 32 + lhalf * 16];
      sc0 = __builtin_amdgcn_wmma_f32_16x16x32_f16(
          false, qf[ks], false, b0, (short)0, sc0, false, false);
      const v16h b1 = *(const v16h*)&Ks[(16 + lrow) * 64 + ks * 32 + lhalf * 16];
      sc1 = __builtin_amdgcn_wmma_f32_16x16x32_f16(
          false, qf[ks], false, b1, (short)0, sc1, false, false);
    }

    const int key0 = kt * 32 + lrow;
    const float add0 = token_valid(key0, ld, lt)      ? 0.f : -1e30f;
    const float add1 = token_valid(key0 + 16, ld, lt) ? 0.f : -1e30f;

    // online softmax: row max via 16-lane butterfly; sums via WMMA below
    float alpha[8], p0[8], p1[8];
    #pragma unroll
    for (int r = 0; r < 8; ++r) {
      const float s0 = sc0[r] * scale + add0;
      const float s1 = sc1[r] * scale + add1;
      float mx = fmaxf(s0, s1);
      #pragma unroll
      for (int off = 1; off < 16; off <<= 1) mx = fmaxf(mx, __shfl_xor(mx, off, 32));
      const float mnew = fmaxf(mrow[r], mx);
      alpha[r] = __expf(mrow[r] - mnew);
      p0[r] = __expf(s0 - mnew);
      p1[r] = __expf(s1 - mnew);
      mrow[r] = mnew;
    }
    #pragma unroll
    for (int nt = 0; nt < 4; ++nt)
      #pragma unroll
      for (int r = 0; r < 8; ++r) o[nt][r] *= alpha[r];
    #pragma unroll
    for (int r = 0; r < 8; ++r) lacc[r] *= alpha[r];

    // C-layout -> A-layout via per-wave LDS
    #pragma unroll
    for (int r = 0; r < 8; ++r) {
      const int mq = r + lhalf * 8;
      Ps[wv][mq * 32 + lrow]      = (_Float16)p0[r];
      Ps[wv][mq * 32 + 16 + lrow] = (_Float16)p1[r];
    }
    __syncthreads();

    const _Float16* pp = &Ps[wv][lrow * 32 + lhalf * 16];
    const v8h pa = *(const v8h*)(pp);
    const v8h pb = *(const v8h*)(pp + 8);
    const v16h pf = __builtin_shufflevector(pa, pb,
        0, 1, 2, 3, 4, 5, 6, 7, 8, 9, 10, 11, 12, 13, 14, 15);

    // normalizer accumulation: every column of lacc = row-sum of P
    lacc = __builtin_amdgcn_wmma_f32_16x16x32_f16(
        false, pf, false, onesf, (short)0, lacc, false, false);

    #pragma unroll
    for (int nt = 0; nt < 4; ++nt) {
      const v16h bv = *(const v16h*)&Vt[(nt * 16 + lrow) * 32 + lhalf * 16];
      o[nt] = __builtin_amdgcn_wmma_f32_16x16x32_f16(
          false, pf, false, bv, (short)0, o[nt], false, false);
    }
    __syncthreads();
  }

  #pragma unroll
  for (int r = 0; r < 8; ++r) {
    const float inv = 1.0f / lacc[r];
    const int q = q0 + wv * 16 + r + lhalf * 8;
    #pragma unroll
    for (int nt = 0; nt < 4; ++nt) {
      Oh[base + (size_t)q * kD + nt * 16 + lrow] = (_Float16)(o[nt][r] * inv);
    }
  }
}

// ---------------------------------------------------------------------------
// X = LayerNorm(X + Y) * (mask?)  -> writes X f32 and Xh f16
// ---------------------------------------------------------------------------
__global__ void __launch_bounds__(256)
add_ln_kernel(float* __restrict__ X, _Float16* __restrict__ Xh,
              const float* __restrict__ Y,
              const float* __restrict__ lw, const float* __restrict__ lb,
              const int* __restrict__ lend, const int* __restrict__ lent,
              int apply_mask) {
  __shared__ float red[8];
  const int t = blockIdx.x;
  const int b = t / kS;
  const int s = t % kS;

  float vals[4];
  float sum = 0.f;
  #pragma unroll
  for (int i = 0; i < 4; ++i) {
    const int d = threadIdx.x + i * 256;
    vals[i] = X[(size_t)t * kD + d] + Y[(size_t)t * kD + d];
    sum += vals[i];
  }
  const float mean = block_sum_256(sum, red) * (1.0f / kD);
  float vs = 0.f;
  #pragma unroll
  for (int i = 0; i < 4; ++i) { const float c = vals[i] - mean; vs += c * c; }
  const float var = block_sum_256(vs, red) * (1.0f / kD);
  const float inv = rsqrtf(var + 1e-12f);
  float msk = 1.0f;
  if (apply_mask) msk = token_valid(s, lend[b], lent[b]) ? 1.0f : 0.0f;
  #pragma unroll
  for (int i = 0; i < 4; ++i) {
    const int d = threadIdx.x + i * 256;
    const float o = ((vals[i] - mean) * inv * lw[d] + lb[d]) * msk;
    X[(size_t)t * kD + d]  = o;
    Xh[(size_t)t * kD + d] = (_Float16)o;
  }
}

// ---------------------------------------------------------------------------
__global__ void __launch_bounds__(256)
transpose_out_kernel(const float* __restrict__ X, float* __restrict__ out) {
  const int t = blockIdx.x;
  const int b = t / kS;
  const int s = t % kS;
  #pragma unroll
  for (int i = 0; i < 4; ++i) {
    const int d = threadIdx.x + i * 256;
    out[((size_t)s * kB + b) * kD + d] = X[(size_t)t * kD + d];
  }
}

// ---------------------------------------------------------------------------
extern "C" void kernel_launch(void* const* d_in, const int* in_sizes, int n_in,
                              void* d_out, int out_size, void* d_ws, size_t ws_size,
                              hipStream_t stream) {
  (void)in_sizes; (void)n_in; (void)out_size; (void)ws_size;

  const float* x_data   = (const float*)d_in[0];
  const float* x_text   = (const float*)d_in[1];
  const float* type_emb = (const float*)d_in[2];
  const float* ln_emb_w = (const float*)d_in[3];
  const float* ln_emb_b = (const float*)d_in[4];
  const float* Wq = (const float*)d_in[5];
  const float* bq = (const float*)d_in[6];
  const float* Wk = (const float*)d_in[7];
  const float* bk = (const float*)d_in[8];
  const float* Wv = (const float*)d_in[9];
  const float* bv = (const float*)d_in[10];
  const float* Wo = (const float*)d_in[11];
  const float* bo = (const float*)d_in[12];
  const float* W1 = (const float*)d_in[13];
  const float* b1 = (const float*)d_in[14];
  const float* W2 = (const float*)d_in[15];
  const float* b2 = (const float*)d_in[16];
  const float* l1w = (const float*)d_in[17];
  const float* l1b = (const float*)d_in[18];
  const float* l2w = (const float*)d_in[19];
  const float* l2b = (const float*)d_in[20];
  const int* lend = (const int*)d_in[21];
  const int* lent = (const int*)d_in[22];

  // ---- workspace carve ----
  const size_t nTD  = (size_t)kTOK * kD;    // 4M
  const size_t nTF  = (size_t)kTOK * kDFF;  // 16M
  const size_t nWdd = (size_t)6 * kD * kD;  // 6M
  const size_t nWfd = (size_t)6 * kDFF * kD;// 24M

  float*    X   = (float*)d_ws;
  float*    Yb  = X + nTD;
  _Float16* Xh  = (_Float16*)(Yb + nTD);
  _Float16* Qh  = Xh + nTD;
  _Float16* Kh  = Qh + nTD;
  _Float16* Vh  = Kh + nTD;
  _Float16* Ah  = Vh + nTD;
  _Float16* Hh  = Ah + nTD;
  _Float16* Wqh = Hh + nTF;
  _Float16* Wkh = Wqh + nWdd;
  _Float16* Wvh = Wkh + nWdd;
  _Float16* Woh = Wvh + nWdd;
  _Float16* W1h = Woh + nWdd;
  _Float16* W2h = W1h + nWfd;

  // ---- per-call weight conversion to f16 ----
  f32_to_f16_kernel<<<dim3((unsigned)(nWdd / 1024)), 256, 0, stream>>>(Wq, Wqh);
  f32_to_f16_kernel<<<dim3((unsigned)(nWdd / 1024)), 256, 0, stream>>>(Wk, Wkh);
  f32_to_f16_kernel<<<dim3((unsigned)(nWdd / 1024)), 256, 0, stream>>>(Wv, Wvh);
  f32_to_f16_kernel<<<dim3((unsigned)(nWdd / 1024)), 256, 0, stream>>>(Wo, Woh);
  f32_to_f16_kernel<<<dim3((unsigned)(nWfd / 1024)), 256, 0, stream>>>(W1, W1h);
  f32_to_f16_kernel<<<dim3((unsigned)(nWfd / 1024)), 256, 0, stream>>>(W2, W2h);

  const dim3 blk256(256), blk128(128);
  const dim3 gProj(kD / 128, kTOK / 128);
  const dim3 gFF1(kDFF / 128, kTOK / 128);
  const dim3 gAttn(kS / 64, kNH, kB);

  embed_kernel<<<kTOK, blk256, 0, stream>>>(x_data, x_text, type_emb,
                                            ln_emb_w, ln_emb_b, lend, lent, X, Xh);

  for (int l = 0; l < 6; ++l) {
    const _Float16* wq = Wqh + (size_t)l * kD * kD;
    const _Float16* wk = Wkh + (size_t)l * kD * kD;
    const _Float16* wv = Wvh + (size_t)l * kD * kD;
    const _Float16* wo = Woh + (size_t)l * kD * kD;
    const _Float16* w1 = W1h + (size_t)l * kDFF * kD;
    const _Float16* w2 = W2h + (size_t)l * kD * kDFF;

    gemm_wmma<1><<<gProj, blk256, 0, stream>>>(Xh, wq, bq + l * kD, Qh, kTOK, kD, kD);
    gemm_wmma<1><<<gProj, blk256, 0, stream>>>(Xh, wk, bk + l * kD, Kh, kTOK, kD, kD);
    gemm_wmma<1><<<gProj, blk256, 0, stream>>>(Xh, wv, bv + l * kD, Vh, kTOK, kD, kD);

    attention_kernel<<<gAttn, blk128, 0, stream>>>(Qh, Kh, Vh, Ah, lend, lent);

    gemm_wmma<0><<<gProj, blk256, 0, stream>>>(Ah, wo, bo + l * kD, Yb, kTOK, kD, kD);
    add_ln_kernel<<<kTOK, blk256, 0, stream>>>(X, Xh, Yb, l1w + l * kD, l1b + l * kD,
                                               lend, lent, 0);

    gemm_wmma<2><<<gFF1, blk256, 0, stream>>>(Xh, w1, b1 + (size_t)l * kDFF, Hh,
                                              kTOK, kDFF, kD);
    gemm_wmma<0><<<gProj, blk256, 0, stream>>>(Hh, w2, b2 + l * kD, Yb,
                                               kTOK, kD, kDFF);
    add_ln_kernel<<<kTOK, blk256, 0, stream>>>(X, Xh, Yb, l2w + l * kD, l2b + l * kD,
                                               lend, lent, 1);
  }

  transpose_out_kernel<<<kTOK, blk256, 0, stream>>>(X, (float*)d_out);
}
